// PoincareConcatLinear_56203942035756
// MI455X (gfx1250) — matrix operand order repaired and checked by reading, not compiled, not run on hardware
//
#include <hip/hip_runtime.h>
#include <hip/hip_bf16.h>
#include <math.h>

typedef __attribute__((ext_vector_type(16))) _Float16 v16h;
typedef __attribute__((ext_vector_type(8)))  _Float16 v8h;
typedef __attribute__((ext_vector_type(8)))  float    v8f;

#define IN_STACKS 4
#define KDIM      1024
#define NDIM      1024
#define M_BLK     32
#define LDA       1032            // halves per A-tile row (pad -> conflict-free ds_load_b128)
#define RC        1.0f            // sqrt(C), C = 1
#define BETA_RATIO 0.4996339f     // beta(512,.5)/beta(128,.5)
#define EPS_PROJ  4e-3f
#define MIN_NORM  1e-15f

// ---------------------------------------------------------------------------
// Kernel 0a: per-output-column reciprocal norm of weight_v + per-column
// epilogue constants cosh/sinh(2*rc*bias), 2*g/rc.
// ---------------------------------------------------------------------------
__global__ __launch_bounds__(256) void prep_cols(
    const float* __restrict__ wv, const float* __restrict__ wg,
    const float* __restrict__ bias, float* __restrict__ rnorm,
    float* __restrict__ coshd, float* __restrict__ sinhd,
    float* __restrict__ gsc)
{
  __shared__ float red[4][64];
  const int t  = threadIdx.x;
  const int nl = t & 63;
  const int kq = t >> 6;
  const int n  = blockIdx.x * 64 + nl;

  float s = 0.f;
  for (int k = kq; k < KDIM; k += 4) {
    float v = wv[(size_t)k * NDIM + n];
    s += v * v;
  }
  red[kq][nl] = s;
  __syncthreads();
  if (kq == 0) {
    float tot = red[0][nl] + red[1][nl] + red[2][nl] + red[3][nl];
    rnorm[n] = 1.0f / fmaxf(sqrtf(tot), MIN_NORM);
    float d  = 2.0f * RC * bias[n];
    coshd[n] = coshf(d);
    sinhd[n] = sinhf(d);
    gsc[n]   = 2.0f * wg[n] / RC;
  }
}

// ---------------------------------------------------------------------------
// Kernel 0b: zuT[n][k] = (f16)(wv[k][n] * rnorm[n])  (LDS-tiled transpose so
// WMMA B fragments are contiguous 32-byte loads per lane).
// ---------------------------------------------------------------------------
__global__ __launch_bounds__(256) void prep_zuT(
    const float* __restrict__ wv, const float* __restrict__ rnorm,
    _Float16* __restrict__ zuT)
{
  __shared__ float tile[64][65];
  const int t  = threadIdx.x;
  const int j  = t & 63;
  const int i0 = t >> 6;
  const int k0 = blockIdx.x * 64;
  const int n0 = blockIdx.y * 64;

#pragma unroll
  for (int ii = 0; ii < 16; ++ii) {
    int i = i0 + ii * 4;
    tile[i][j] = wv[(size_t)(k0 + i) * NDIM + (n0 + j)];
  }
  __syncthreads();
#pragma unroll
  for (int ii = 0; ii < 16; ++ii) {
    int i = i0 + ii * 4;
    float rn = rnorm[n0 + i];
    zuT[(size_t)(n0 + i) * KDIM + (k0 + j)] = (_Float16)(tile[j][i] * rn);
  }
}

// ---------------------------------------------------------------------------
// Main fused kernel: one block = 32 rows x all 1024 outputs (8 waves, each
// wave: 2 M-tiles x 8 N-tiles of v_wmma_f32_16x16x32_f16).
// Epilogue identity: sinh(g*asinh(q)) = (w^g - w^-g)/2, w = q + sqrt(q^2+1),
// computed with raw v_log_f32 / v_exp_f32 / v_rcp_f32 / v_sqrt_f32.
// ---------------------------------------------------------------------------
__global__ __launch_bounds__(256) void poincare_main(
    const float* __restrict__ x, const _Float16* __restrict__ zuT,
    const float* __restrict__ coshd, const float* __restrict__ sinhd,
    const float* __restrict__ gsc, float* __restrict__ out)
{
  __shared__ _Float16 a_tile[M_BLK * LDA];
  __shared__ float part[M_BLK][8];
  __shared__ float coefs[M_BLK][IN_STACKS];
  __shared__ float opcs[M_BLK];     // 1 + cx2
  __shared__ float rdens[M_BLK];    // 1 / max(1 - cx2, 1e-15)
  __shared__ float rowsum[M_BLK];
  __shared__ float rowscale[M_BLK];

  const int tid  = threadIdx.x;
  const int row0 = blockIdx.x * M_BLK;
  const int r    = tid >> 3;     // 0..31  row within block
  const int seg  = tid & 7;      // 0..7   128-elem segment within row

  // ---- stage 1: partial sum of squares per (row, segment) ----
  const float* xrow = x + (size_t)(row0 + r) * KDIM + seg * 128;
  {
    float s = 0.f;
#pragma unroll
    for (int j2 = 0; j2 < 128; j2 += 4) {
      float4 v = *(const float4*)(xrow + j2);
      s += v.x * v.x + v.y * v.y + v.z * v.z + v.w * v.w;
    }
    part[r][seg] = s;
  }
  __syncthreads();

  if (seg == 0) {
    // logmap0 per stack, concat, *beta ratio, expmap0, project
    float atsq = 0.f;
    float al[IN_STACKS];
#pragma unroll
    for (int st = 0; st < IN_STACKS; ++st) {
      float n2 = part[r][2*st] + part[r][2*st + 1];
      float nn = fmaxf(sqrtf(n2), MIN_NORM);
      float tc = fminf(RC * nn, 1.0f - 1e-7f);
      float at = atanhf(tc);
      al[st] = at / (RC * nn);
      atsq  += at * at;
    }
    float un  = fmaxf(BETA_RATIO * sqrtf(atsq) / RC, MIN_NORM);
    float th  = tanhf(RC * un);
    float gg  = th / (RC * un);              // expmap0 direction scale
    float hn  = th / RC;                     // ||h|| before projection
    float mx  = (1.0f - EPS_PROJ) / RC;
    float clip = (hn > mx) ? (mx / hn) : 1.0f;
    float hcl  = fminf(hn, mx);
    float cx2  = (RC * hcl) * (RC * hcl);
    opcs[r]  = 1.0f + cx2;
    rdens[r] = 1.0f / fmaxf(1.0f - cx2, 1e-15f);
    float base = RC * clip * gg * BETA_RATIO;   // folds rc into A tile
#pragma unroll
    for (int st = 0; st < IN_STACKS; ++st) coefs[r][st] = base * al[st];
  }
  if (tid < M_BLK) rowsum[tid] = 0.f;
  __syncthreads();

  // ---- stage 2: scaled f16 A tile (x rows hot in cache from stage 1) ----
  {
    float c0 = coefs[r][seg >> 1];           // 128-seg lies inside one stack
    _Float16* arow = a_tile + r * LDA + seg * 128;
#pragma unroll
    for (int j2 = 0; j2 < 128; j2 += 4) {
      float4 v = *(const float4*)(xrow + j2);
      arow[j2 + 0] = (_Float16)(c0 * v.x);
      arow[j2 + 1] = (_Float16)(c0 * v.y);
      arow[j2 + 2] = (_Float16)(c0 * v.z);
      arow[j2 + 3] = (_Float16)(c0 * v.w);
    }
  }
  __syncthreads();

  // ---- stage 3: WMMA GEMM, 2 M-tiles x 8 N-tiles per wave ----
  const int lane   = tid & 31;
  const int wid    = tid >> 5;
  const int l      = lane & 15;
  const int hi     = lane >> 4;
  const int n_base = wid * 128;

  v8f acc[2][8];
  const v8f vzero = {0.f, 0.f, 0.f, 0.f, 0.f, 0.f, 0.f, 0.f};
#pragma unroll
  for (int mt = 0; mt < 2; ++mt)
#pragma unroll
    for (int nt = 0; nt < 8; ++nt) acc[mt][nt] = vzero;

  const _Float16* arow0 = a_tile + l * LDA;
  const _Float16* arow1 = a_tile + (l + 16) * LDA;
  for (int k0 = 0; k0 < KDIM; k0 += 32) {
    // A operand in contiguous k order (same as B): lane(l,hi): elems 0..15 = K[k0+16hi..k0+16hi+15]
    v8h a0lo = *(const v8h*)(arow0 + k0 + hi * 16);
    v8h a0hi = *(const v8h*)(arow0 + k0 + hi * 16 + 8);
    v16h af0 = __builtin_shufflevector(a0lo, a0hi,
        0, 1, 2, 3, 4, 5, 6, 7, 8, 9, 10, 11, 12, 13, 14, 15);
    v8h a1lo = *(const v8h*)(arow1 + k0 + hi * 16);
    v8h a1hi = *(const v8h*)(arow1 + k0 + hi * 16 + 8);
    v16h af1 = __builtin_shufflevector(a1lo, a1hi,
        0, 1, 2, 3, 4, 5, 6, 7, 8, 9, 10, 11, 12, 13, 14, 15);
#pragma unroll
    for (int nt = 0; nt < 8; ++nt) {
      int n = n_base + nt * 16 + l;
      // B operand, same contiguous k order as A: lane(l,hi) elem i = B[k0 + 16*hi + i][n] -> contiguous in zuT
      v16h bf = *(const v16h*)(zuT + (size_t)n * KDIM + k0 + hi * 16);
      acc[0][nt] = __builtin_amdgcn_wmma_f32_16x16x32_f16(
          false, af0, false, bf, (short)0, acc[0][nt], false, false);
      acc[1][nt] = __builtin_amdgcn_wmma_f32_16x16x32_f16(
          false, af1, false, bf, (short)0, acc[1][nt], false, false);
    }
  }

  // ---- epilogue: MLR + sinh via w^g identity; y2 overwrites acc in place ----
  float rs[2][8];
#pragma unroll
  for (int mt = 0; mt < 2; ++mt)
#pragma unroll
    for (int j2 = 0; j2 < 8; ++j2) rs[mt][j2] = 0.f;

#pragma unroll
  for (int nt = 0; nt < 8; ++nt) {
    int n = n_base + nt * 16 + l;
    float ch2 = 2.0f * coshd[n];
    float sh  = sinhd[n];
    float gv  = gsc[n];
#pragma unroll
    for (int mt = 0; mt < 2; ++mt) {
#pragma unroll
      for (int j2 = 0; j2 < 8; ++j2) {
        int   m   = mt * 16 + hi * 8 + j2;    // C/D layout: VGPR j -> M = j + 8*hi
        float mm  = acc[mt][nt][j2];          // = rc*h . z_unit_n
        float num = __builtin_fmaf(mm, ch2, -(opcs[m] * sh));
        float q   = num * rdens[m];
        float w   = q + __builtin_amdgcn_sqrtf(__builtin_fmaf(q, q, 1.0f));
        float t   = __builtin_amdgcn_logf(w);            // log2(w), w > 0 always
        float p   = __builtin_amdgcn_exp2f(gv * t);      // w^g  (rc = 1)
        float y2  = 0.5f * (p - __builtin_amdgcn_rcpf(p));
        acc[mt][nt][j2] = y2;
        rs[mt][j2] = __builtin_fmaf(y2, y2, rs[mt][j2]);
      }
    }
  }
#pragma unroll
  for (int mt = 0; mt < 2; ++mt)
#pragma unroll
    for (int j2 = 0; j2 < 8; ++j2)
      atomicAdd(&rowsum[mt * 16 + hi * 8 + j2], rs[mt][j2]);
  __syncthreads();

  if (tid < M_BLK) {
    float S     = rowsum[tid];
    float denom = 1.0f + sqrtf(1.0f + S);            // c = 1
    float nrm   = fmaxf(sqrtf(S), MIN_NORM) / denom;
    float mx    = (1.0f - EPS_PROJ) / RC;
    float sc    = (nrm > mx) ? (mx / nrm) : 1.0f;
    rowscale[tid] = sc / denom;
  }
  __syncthreads();

#pragma unroll
  for (int nt = 0; nt < 8; ++nt) {
    int n = n_base + nt * 16 + l;
#pragma unroll
    for (int mt = 0; mt < 2; ++mt) {
#pragma unroll
      for (int j2 = 0; j2 < 8; ++j2) {
        int m = mt * 16 + hi * 8 + j2;
        out[(size_t)(row0 + m) * NDIM + n] = acc[mt][nt][j2] * rowscale[m];
      }
    }
  }
}

// ---------------------------------------------------------------------------
extern "C" void kernel_launch(void* const* d_in, const int* in_sizes, int n_in,
                              void* d_out, int out_size, void* d_ws, size_t ws_size,
                              hipStream_t stream) {
  (void)in_sizes; (void)n_in; (void)out_size; (void)ws_size;
  const float* x    = (const float*)d_in[0];   // [16384, 4, 256]
  const float* wg   = (const float*)d_in[1];   // [1024]
  const float* wv   = (const float*)d_in[2];   // [1024, 1024]
  const float* bias = (const float*)d_in[3];   // [1024]
  float* out = (float*)d_out;                  // [16384, 1024]

  char* ws = (char*)d_ws;
  _Float16* zuT   = (_Float16*)ws;                          // 2 MB
  float*    rnorm = (float*)(ws + (size_t)KDIM * NDIM * 2); // 4 KB
  float*    coshd = rnorm + NDIM;
  float*    sinhd = coshd + NDIM;
  float*    gsc   = sinhd + NDIM;

  prep_cols<<<NDIM / 64, 256, 0, stream>>>(wv, wg, bias, rnorm, coshd, sinhd, gsc);
  prep_zuT<<<dim3(KDIM / 64, NDIM / 64), 256, 0, stream>>>(wv, rnorm, zuT);
  poincare_main<<<16384 / M_BLK, 256, 0, stream>>>(x, zuT, coshd, sinhd, gsc, out);
}
